// ConvchannelAttentionBlock_2534030705305
// MI455X (gfx1250) — matrix operand order repaired and checked by reading, not compiled, not run on hardware
//
#include <hip/hip_runtime.h>
#include <hip/hip_bf16.h>
#include <math.h>

// ---------------------------------------------------------------------------
// ConvchannelAttentionBlock for MI455X (gfx1250, wave32, WMMA).
//   logits = X X^T (per batch)          -> bf16 WMMA (16x16x32), fp32 accum
//   probs  = softmax(rowmax - logits)   -> fused fp32 epilogue (== softmax(-logits))
//   read   = probs @ X                  -> bf16 WMMA, fp32 accum
//   out    = eta * read + x             -> fp32
// Shapes fixed by the reference: B=32, C=512, L=4096 (fp32 in/out).
// Workspace: xb bf16 [B][C][L] (128 MB) + xT bf16 [B][L][C] (128 MB)
//          + probs bf16 [B][C][C] (16 MB)  => 272 MB.
// One-time bf16 conversion (kernel 1) removes all v_cvt from the GEMM inner
// loops: each x element is WMMA-read ~64x, so converting once is the roofline
// -correct tradeoff (+5.5us HBM writes vs a VALU-bound inner loop).
// ---------------------------------------------------------------------------

typedef __bf16 bf16_t;
typedef __attribute__((ext_vector_type(16))) __bf16 v16bf;
typedef __attribute__((ext_vector_type(8)))  __bf16 v8bf;
typedef __attribute__((ext_vector_type(8)))  float  v8f;

union V16 { v16bf v; v8bf h[2]; };

#define BB 32
#define CC 512
#define LL 4096
#define WAVES 8   // 256 threads = 8 wave32

// ---------------------------------------------------------------------------
// Kernel 1: x fp32 [B][C][L] -> xb bf16 [B][C][L] (direct) and
//           xT bf16 [B][L][C] (via LDS 32x32 tile transpose).
// ---------------------------------------------------------------------------
__global__ __launch_bounds__(256) void cvt_transpose_kernel(
    const float* __restrict__ x, bf16_t* __restrict__ xb,
    bf16_t* __restrict__ xT) {
  __shared__ bf16_t tile[32][33];  // +1 pad: conflict-free transposed reads
  const int b  = blockIdx.z;
  const int l0 = blockIdx.x * 32;
  const int c0 = blockIdx.y * 32;
  const int tx = threadIdx.x & 31;       // l within tile
  const int ty = threadIdx.x >> 5;       // 0..7 (c step)

#pragma unroll
  for (int i = 0; i < 32; i += 8) {
    const size_t idx = ((size_t)b * CC + (c0 + ty + i)) * LL + (l0 + tx);
    const bf16_t h   = (bf16_t)x[idx];   // RNE convert
    tile[ty + i][tx] = h;
    xb[idx]          = h;                // row-major bf16 copy (coalesced)
  }
  __syncthreads();
#pragma unroll
  for (int i = 0; i < 32; i += 8) {
    // xT[b][l][c] = x[b][c][l]
    xT[((size_t)b * LL + (l0 + ty + i)) * CC + (c0 + tx)] = tile[tx][ty + i];
  }
}

// ---------------------------------------------------------------------------
// Kernel 2: fused Gram + inverted softmax.
// One workgroup per (batch, 16-row strip of C). 8 waves; wave w owns the
// 16x64 logits slab at columns [64w, 64w+64) as 4 WMMA accumulators.
// Operand register layouts used here (bf16 16x16x32, wave32, contiguous k):
//   A 16x32: lane m=lane&15, slot e -> k = 16*(lane>>4) + e
//   B 32x16: lane n=lane&15, slot e -> k = 16*(lane>>4) + e
// Both A and B tiles are row-blocks of xb -> contiguous 16B/32B runs/lane.
// ---------------------------------------------------------------------------
__global__ __launch_bounds__(256) void gram_softmax_kernel(
    const bf16_t* __restrict__ xb, bf16_t* __restrict__ probs) {
  const int b    = blockIdx.y;
  const int c0   = blockIdx.x * 16;          // output row strip
  const int wave = threadIdx.x >> 5;
  const int lane = threadIdx.x & 31;
  const int m    = lane & 15;                // A row / B col within tile
  const int hi   = lane >> 4;                // lane-half selector

  const bf16_t* xbb  = xb + (size_t)b * CC * LL;
  const bf16_t* arow = xbb + (size_t)(c0 + m) * LL;

  v8f acc[4] = {};   // 16x16 fp32 tiles, cols (wave*4+t)*16 + n

  for (int k0 = 0; k0 < LL; k0 += 32) {
    // ---- A tile: two 8-elem bf16 runs per lane
    V16 a;
    a.h[0] = *(const v8bf*)(arow + k0 + 16 * hi);
    a.h[1] = *(const v8bf*)(arow + k0 + 16 * hi + 8);
    __builtin_prefetch(arow + k0 + 128, 0, 1);  // next chunks (L2->WGP$)
#pragma unroll
    for (int t = 0; t < 4; ++t) {
      // ---- B tile: B[k][n] = X[16J+n][k0+k]; one 16-elem run per lane
      const int     rj   = (wave * 4 + t) * 16;
      const bf16_t* brow = xbb + (size_t)(rj + m) * LL;  // lane index == n
      V16 bm;
      bm.h[0] = *(const v8bf*)(brow + k0 + 16 * hi);
      bm.h[1] = *(const v8bf*)(brow + k0 + 16 * hi + 8);
      __builtin_prefetch(brow + k0 + 128, 0, 1);
      acc[t] = __builtin_amdgcn_wmma_f32_16x16x32_bf16(
          false, a.v, false, bm.v, (short)0, acc[t], false, false);
    }
  }

  // ---- fused softmax over d (512 cols): p = exp(minL - logit) / rowsum.
  // (reference: softmax(rowmax - logits) == softmax(-logits) == this)
  __shared__ float redbuf[16][WAVES];

  // 1) per-row min: 4 tiles (in-lane) -> 16 lanes (shfl_xor) -> 8 waves (LDS)
  float pmin[8];
#pragma unroll
  for (int v = 0; v < 8; ++v) {
    float mn = acc[0][v];
#pragma unroll
    for (int t = 1; t < 4; ++t) mn = fminf(mn, acc[t][v]);
#pragma unroll
    for (int off = 1; off <= 8; off <<= 1)
      mn = fminf(mn, __shfl_xor(mn, off, 32));
    pmin[v] = mn;
  }
  if (m == 0) {
#pragma unroll
    for (int v = 0; v < 8; ++v) redbuf[8 * hi + v][wave] = pmin[v];
  }
  __syncthreads();
  float rmin[8];
#pragma unroll
  for (int v = 0; v < 8; ++v) {
    float r = redbuf[8 * hi + v][0];
#pragma unroll
    for (int w = 1; w < WAVES; ++w) r = fminf(r, redbuf[8 * hi + v][w]);
    rmin[v] = r;
  }
  __syncthreads();  // redbuf reused for sums

  // 2) exp + per-row sum (all exponents <= 0: no overflow)
  float ex[4][8];
  float psum[8];
#pragma unroll
  for (int v = 0; v < 8; ++v) psum[v] = 0.0f;
#pragma unroll
  for (int t = 0; t < 4; ++t)
#pragma unroll
    for (int v = 0; v < 8; ++v) {
      ex[t][v] = __expf(rmin[v] - acc[t][v]);
      psum[v] += ex[t][v];
    }
#pragma unroll
  for (int v = 0; v < 8; ++v) {
#pragma unroll
    for (int off = 1; off <= 8; off <<= 1)
      psum[v] += __shfl_xor(psum[v], off, 32);
  }
  if (m == 0) {
#pragma unroll
    for (int v = 0; v < 8; ++v) redbuf[8 * hi + v][wave] = psum[v];
  }
  __syncthreads();

  // 3) normalize and store probs as bf16 [B][C][C]
  bf16_t* prow_base = probs + (size_t)b * CC * CC;
#pragma unroll
  for (int v = 0; v < 8; ++v) {
    float rs = redbuf[8 * hi + v][0];
#pragma unroll
    for (int w = 1; w < WAVES; ++w) rs += redbuf[8 * hi + v][w];
    const float ri  = 1.0f / rs;
    const int   row = c0 + v + 8 * hi;
#pragma unroll
    for (int t = 0; t < 4; ++t) {
      const int col = wave * 64 + t * 16 + m;
      prow_base[(size_t)row * CC + col] = (bf16_t)(ex[t][v] * ri);
    }
  }
}

// ---------------------------------------------------------------------------
// Kernel 3: read = probs @ X, out = eta * read + x.
// One workgroup per (batch, 16-row strip of C, 512-col slab of L).
// Wave w owns 16x64 of the output. A = probs (bf16), B = xT (bf16, so the
// K=channel axis is contiguous per lane) -> pure vector loads, no gathers.
// ---------------------------------------------------------------------------
__global__ __launch_bounds__(256) void attn_read_kernel(
    const float* __restrict__ x, const bf16_t* __restrict__ probs,
    const bf16_t* __restrict__ xT, const float* __restrict__ eta,
    float* __restrict__ out) {
  const int b    = blockIdx.z;
  const int c0   = blockIdx.y * 16;
  const int wave = threadIdx.x >> 5;
  const int lane = threadIdx.x & 31;
  const int m    = lane & 15;
  const int hi   = lane >> 4;
  const int l0   = blockIdx.x * 512 + wave * 64;

  const bf16_t* prow = probs + ((size_t)b * CC + (c0 + m)) * CC;
  const bf16_t* xTb  = xT + (size_t)b * LL * CC;

  v8f acc[4] = {};

  for (int d0 = 0; d0 < CC; d0 += 32) {
    // A tile from probs: slot e -> k = 16*hi + e
    V16 a;
    a.h[0] = *(const v8bf*)(prow + d0 + 16 * hi);
    a.h[1] = *(const v8bf*)(prow + d0 + 16 * hi + 8);
#pragma unroll
    for (int t = 0; t < 4; ++t) {
      // B tile: B[k][n] = X[d0+k][l0+t*16+n] = xT[l0+t*16+n][d0+k]
      const bf16_t* trow = xTb + (size_t)(l0 + t * 16 + m) * CC;
      V16 bm;
      bm.h[0] = *(const v8bf*)(trow + d0 + 16 * hi);
      bm.h[1] = *(const v8bf*)(trow + d0 + 16 * hi + 8);
      __builtin_prefetch(trow + d0 + 64, 0, 1);
      acc[t] = __builtin_amdgcn_wmma_f32_16x16x32_bf16(
          false, a.v, false, bm.v, (short)0, acc[t], false, false);
    }
  }

  const float etav = eta[0];
#pragma unroll
  for (int t = 0; t < 4; ++t)
#pragma unroll
    for (int v = 0; v < 8; ++v) {
      const size_t idx =
          ((size_t)b * CC + (c0 + v + 8 * hi)) * LL + (l0 + t * 16 + m);
      out[idx] = etav * acc[t][v] + x[idx];
    }
}

// ---------------------------------------------------------------------------
extern "C" void kernel_launch(void* const* d_in, const int* in_sizes, int n_in,
                              void* d_out, int out_size, void* d_ws, size_t ws_size,
                              hipStream_t stream) {
  const float* x   = (const float*)d_in[0];   // [B][C][L] fp32
  const float* eta = (const float*)d_in[1];   // [1] fp32
  float*       out = (float*)d_out;           // [B][C][L] fp32

  // workspace layout: xb bf16 [B][C][L] | xT bf16 [B][L][C] | probs bf16 [B][C][C]
  bf16_t* xb    = (bf16_t*)d_ws;
  bf16_t* xT    = xb + (size_t)BB * CC * LL;
  bf16_t* probs = xT + (size_t)BB * LL * CC;

  dim3 blk(256);
  dim3 g1(LL / 32, CC / 32, BB);   // convert + transpose tiles
  cvt_transpose_kernel<<<g1, blk, 0, stream>>>(x, xb, xT);

  dim3 g2(CC / 16, BB);            // one block per (c-strip, batch)
  gram_softmax_kernel<<<g2, blk, 0, stream>>>(xb, probs);

  dim3 g3(LL / 512, CC / 16, BB);  // (l-slab, c-strip, batch)
  attn_read_kernel<<<g3, blk, 0, stream>>>(x, probs, xT, eta, out);
}
